// LTAE2d_43044162240863
// MI455X (gfx1250) — matrix operand order repaired and checked by reading, not compiled, not run on hardware
//
#include <hip/hip_runtime.h>
#include <math.h>

typedef __attribute__((ext_vector_type(16))) _Float16 v16h;
typedef __attribute__((ext_vector_type(8)))  float    v8f;

#define GN_EPS 1e-5f

// ---------------------------------------------------------------------------
// WMMA helpers (CDNA5 gfx1250, wave32): D = A(16x32 f16) * B(32x16 f16) + C(f32)
// ---------------------------------------------------------------------------
__device__ __forceinline__ v8f wmma16(v16h a, v16h b, v8f c) {
  // 8 args: (neg_a, A, neg_b, B, c_mod, C, reuse_a, reuse_b)
  return __builtin_amdgcn_wmma_f32_16x16x32_f16(false, a, false, b, (short)0, c,
                                                false, false);
}

// A fragment: 16x32 f16 tile from LDS (row-major, strideH halves per row).
// ISA layout: lane<16 holds row=lane, K in {kb+0..7, kb+16..23};
//             lane>=16 holds row=lane-16, K in {kb+8..15, kb+24..31}.
// => two contiguous 16B chunks per lane.
__device__ __forceinline__ v16h lds_a_frag(const _Float16* base, int strideH,
                                           int kbase, int lane) {
  int row = lane & 15;
  int b0  = kbase + ((lane >> 4) << 3);
  const _Float16* p = base + row * strideH + b0;
  union { v16h h; uint4 q[2]; } u;
  u.q[0] = *(const uint4*)(p);
  u.q[1] = *(const uint4*)(p + 16);
  return u.h;
}

// B fragment: B[k][n] = W[n][k] with W (out x in) row-major f16 in global.
// lane holds column n = lane&15, K in {kb+8h..+7, kb+16+8h..+7} (h = lane>>4) => two 16B chunks.
__device__ __forceinline__ v16h glb_b_frag(const _Float16* w, int K, int n0,
                                           int kbase, int lane) {
  int n = n0 + (lane & 15);
  int k = kbase + ((lane >> 4) << 3);
  const uint4* p = (const uint4*)(w + (size_t)n * K + k);
  union { v16h h; uint4 q[2]; } u;
  u.q[0] = p[0];
  u.q[1] = p[2];
  return u.h;
}

// Column-group reduction across lanes that share the same C-matrix rows.
// MAXM=8 -> reduce over a 16-lane half (full 16-column span per mask set),
// MAXM=4 -> reduce over aligned 8-lane group (8 columns).
template <int MAXM>
__device__ __forceinline__ void reduce_cols(float s[8], float q[8]) {
#pragma unroll
  for (int m = 1; m <= MAXM; m <<= 1) {
#pragma unroll
    for (int e = 0; e < 8; ++e) {
      s[e] += __shfl_xor(s[e], m, 32);
      q[e] += __shfl_xor(q[e], m, 32);
    }
  }
}

// ---------------------------------------------------------------------------
// Kernel 0: convert the four MLP weight matrices fp32 -> f16 into workspace
// layout: [w_in0 32768][w_in1 32768][w_m0 32768][w_m1 32768]
// ---------------------------------------------------------------------------
__global__ __launch_bounds__(256) void cvt_w(const float* __restrict__ w0,
                                             const float* __restrict__ w1,
                                             const float* __restrict__ m0,
                                             const float* __restrict__ m1,
                                             _Float16* __restrict__ out) {
  int i = blockIdx.x * 256 + threadIdx.x;  // 131072 total
  float v;
  if (i < 32768)        v = w0[i];
  else if (i < 65536)   v = w1[i - 32768];
  else if (i < 98304)   v = m0[i - 65536];
  else                  v = m1[i - 98304];
  out[i] = (_Float16)v;
}

// ---------------------------------------------------------------------------
// Kernel 1: inconv MLP  (per token: 128 -> GN4+relu(256) -> GN4+relu(128))
// One wave per tile of 16 tokens (16 consecutive w at fixed b,t,h).
// Output ws1 layout: (n, t, c) fp32 with n = (b*48+h)*48+w.
// ---------------------------------------------------------------------------
__global__ __launch_bounds__(128) void ltae_inconv(
    const float* __restrict__ x,
    const _Float16* __restrict__ w0f, const float* __restrict__ b0,
    const float* __restrict__ g0, const float* __restrict__ be0,
    const _Float16* __restrict__ w1f, const float* __restrict__ b1,
    const float* __restrict__ g1, const float* __restrict__ be1,
    float* __restrict__ ws1) {
  __shared__ __align__(16) _Float16 ldsA[4][16 * 136];  // input tile  f16
  __shared__ __align__(16) _Float16 ldsB[4][16 * 264];  // hidden tile f16 / f32 staging
  const int wave = threadIdx.x >> 5;
  const int lane = threadIdx.x & 31;
  const int half = lane >> 4;
  const int col  = lane & 15;

  const int tt  = blockIdx.x * 4 + wave;  // 13824 tiles total
  const int wt  = tt % 3;
  const int bth = tt / 3;
  const int hh  = bth % 48;
  const int bt  = bth / 48;
  const int t   = bt % 24;
  const int b   = bt / 24;
  const int w0c = wt * 16;

  _Float16* A  = ldsA[wave];
  _Float16* Bh = ldsB[wave];
  float*    Sf = (float*)ldsB[wave];

  // ---- stage 16 tokens x 128 channels (gather over c, coalesced over w) ----
  const float* xb =
      x + ((size_t)(b * 24 + t)) * 128 * 2304 + (size_t)hh * 48 + w0c;
#pragma unroll 4
  for (int it = 0; it < 64; ++it) {
    int idx = it * 32 + lane;
    int c = idx >> 4, wv = idx & 15;
    A[wv * 136 + c] = (_Float16)xb[(size_t)c * 2304 + wv];
  }

  v16h afr[4];
#pragma unroll
  for (int kc = 0; kc < 4; ++kc) afr[kc] = lds_a_frag(A, 136, kc * 32, lane);

  // ---- layer 0: 128 -> 256, GN(4) over 64-channel groups (4 tiles), relu ---
#pragma unroll
  for (int g = 0; g < 4; ++g) {
    v8f acc[4];
#pragma unroll
    for (int jj = 0; jj < 4; ++jj) {
      int j = g * 4 + jj;
      v8f c8 = {0.f, 0.f, 0.f, 0.f, 0.f, 0.f, 0.f, 0.f};
#pragma unroll
      for (int kc = 0; kc < 4; ++kc)
        c8 = wmma16(afr[kc], glb_b_frag(w0f, 128, j * 16, kc * 32, lane), c8);
      float bias = b0[j * 16 + col];  // bias BEFORE GN (affects stats)
#pragma unroll
      for (int e = 0; e < 8; ++e) c8[e] += bias;
      acc[jj] = c8;
    }
    float s[8], q[8];
#pragma unroll
    for (int e = 0; e < 8; ++e) {
      float sv = 0.f, qv = 0.f;
#pragma unroll
      for (int jj = 0; jj < 4; ++jj) { float v = acc[jj][e]; sv += v; qv += v * v; }
      s[e] = sv; q[e] = qv;
    }
    reduce_cols<8>(s, q);  // sum over 64 columns
#pragma unroll
    for (int jj = 0; jj < 4; ++jj) {
      int cch = (g * 4 + jj) * 16 + col;
      float gam = g0[cch], bet = be0[cch];
#pragma unroll
      for (int e = 0; e < 8; ++e) {
        float mean = s[e] * (1.f / 64.f);
        float var  = q[e] * (1.f / 64.f) - mean * mean;
        float y = (acc[jj][e] - mean) * rsqrtf(var + GN_EPS) * gam + bet;
        y = fmaxf(y, 0.f);
        Bh[(e + 8 * half) * 264 + cch] = (_Float16)y;  // transpose via LDS
      }
    }
  }

  v16h af2[8];
#pragma unroll
  for (int kc = 0; kc < 8; ++kc) af2[kc] = lds_a_frag(Bh, 264, kc * 32, lane);

  // ---- layer 1: 256 -> 128, GN(4) over 32-channel groups (2 tiles), relu ---
#pragma unroll
  for (int g = 0; g < 4; ++g) {
    v8f acc[2];
#pragma unroll
    for (int jj = 0; jj < 2; ++jj) {
      int j = g * 2 + jj;
      v8f c8 = {0.f, 0.f, 0.f, 0.f, 0.f, 0.f, 0.f, 0.f};
#pragma unroll
      for (int kc = 0; kc < 8; ++kc)
        c8 = wmma16(af2[kc], glb_b_frag(w1f, 256, j * 16, kc * 32, lane), c8);
      float bias = b1[j * 16 + col];
#pragma unroll
      for (int e = 0; e < 8; ++e) c8[e] += bias;
      acc[jj] = c8;
    }
    float s[8], q[8];
#pragma unroll
    for (int e = 0; e < 8; ++e) {
      float v0 = acc[0][e], v1 = acc[1][e];
      s[e] = v0 + v1;
      q[e] = v0 * v0 + v1 * v1;
    }
    reduce_cols<8>(s, q);  // sum over 32 columns
#pragma unroll
    for (int jj = 0; jj < 2; ++jj) {
      int cch = (g * 2 + jj) * 16 + col;
      float gam = g1[cch], bet = be1[cch];
#pragma unroll
      for (int e = 0; e < 8; ++e) {
        float mean = s[e] * (1.f / 32.f);
        float var  = q[e] * (1.f / 32.f) - mean * mean;
        float y = (acc[jj][e] - mean) * rsqrtf(var + GN_EPS) * gam + bet;
        y = fmaxf(y, 0.f);
        Sf[(e + 8 * half) * 132 + cch] = y;  // f32 staging (aliases Bh, done with it)
      }
    }
  }

  // ---- coalesced store: one 512B row per iteration ----
  size_t n0 = ((size_t)(b * 48 + hh)) * 48 + w0c;
#pragma unroll 4
  for (int r = 0; r < 16; ++r) {
    float4 v = *(const float4*)(Sf + r * 132 + lane * 4);
    *(float4*)(ws1 + ((n0 + r) * 24 + t) * 128 + lane * 4) = v;
  }
}

// ---------------------------------------------------------------------------
// Kernel 2: per-sequence in_norm GN16 + positional encoding + fixed-query
// attention. One 256-thread block per sequence (9216 blocks).
// ---------------------------------------------------------------------------
__global__ __launch_bounds__(256) void ltae_attn(
    const float* __restrict__ ws1, const int* __restrict__ bpos,
    const float* __restrict__ g_in, const float* __restrict__ b_in,
    const float* __restrict__ Q, const float* __restrict__ w_k,
    const float* __restrict__ b_k, float* __restrict__ ws2) {
  __shared__ float so[24][132];
  __shared__ float sc[16][24];
  __shared__ float posf[24];
  const int n = blockIdx.x;
  const int tid = threadIdx.x;
  const int b = n / 2304;

  const float* src = ws1 + (size_t)n * 3072;
  for (int i = tid; i < 3072; i += 256) so[i >> 7][i & 127] = src[i];
  if (tid < 24) posf[tid] = (float)bpos[b * 24 + tid];
  __syncthreads();

  // GN16 over (8 channels x 24 timesteps), then add sinusoid PE
  {
    int g = tid >> 4, j = tid & 15;  // lanes 0-15 / 16-31 are different groups
    float s = 0.f, q = 0.f;
    for (int e = j; e < 192; e += 16) {
      float v = so[e >> 3][g * 8 + (e & 7)];
      s += v; q += v * v;
    }
    for (int m = 1; m < 16; m <<= 1) {
      s += __shfl_xor(s, m, 32);
      q += __shfl_xor(q, m, 32);
    }
    float mean = s * (1.f / 192.f);
    float var  = q * (1.f / 192.f) - mean * mean;
    float rs = rsqrtf(var + GN_EPS);
    const float invden[4] = {1.f, 1.f / 5.62341325190349f,
                             1.f / 31.6227766016838f, 1.f / 177.827941003892f};
    for (int e = j; e < 192; e += 16) {
      int cl = e & 7, tt = e >> 3, c = g * 8 + cl;
      float v = (so[tt][c] - mean) * rs * g_in[c] + b_in[c];
      float ang = posf[tt] * invden[cl >> 1];
      v += (cl & 1) ? __cosf(ang) : __sinf(ang);
      so[tt][c] = v;
    }
  }
  __syncthreads();

  // scores[h][t] = (1/sqrt(4)) * sum_k Q[h][k] * (b_k + out[t] . w_k_row)
  for (int p = tid; p < 384; p += 256) {
    int h = p / 24, tt = p % 24;
    float scv = 0.f;
#pragma unroll
    for (int kk = 0; kk < 4; ++kk) {
      int row = h * 4 + kk;
      float dot = b_k[row];
      const float* wr = w_k + row * 128;
#pragma unroll 8
      for (int c = 0; c < 128; ++c) dot += so[tt][c] * wr[c];
      scv += Q[h * 4 + kk] * dot;
    }
    sc[h][tt] = scv * 0.5f;
  }
  __syncthreads();

  if (tid < 16) {  // per-head softmax over T=24
    float mx = -3.4e38f;
    for (int tt = 0; tt < 24; ++tt) mx = fmaxf(mx, sc[tid][tt]);
    float ev[24], ss = 0.f;
    for (int tt = 0; tt < 24; ++tt) { ev[tt] = __expf(sc[tid][tt] - mx); ss += ev[tt]; }
    float inv = 1.f / ss;
    for (int tt = 0; tt < 24; ++tt) sc[tid][tt] = ev[tt] * inv;
  }
  __syncthreads();

  if (tid < 128) {  // o[h][v] = sum_t a[h][t] * out[t][8h+v]
    int h = tid >> 3, vv = tid & 7;
    float o = 0.f;
#pragma unroll
    for (int tt = 0; tt < 24; ++tt) o += sc[h][tt] * so[tt][h * 8 + vv];
    ws2[(size_t)n * 128 + tid] = o;
  }
}

// ---------------------------------------------------------------------------
// Kernel 3: output MLP (128->256->128, GN4+relu each) + out_norm GN16,
// scatter to (B, 128, H, W). One wave per 16 sequences; 576 tiles.
// ---------------------------------------------------------------------------
__global__ __launch_bounds__(128) void ltae_outmlp(
    const float* __restrict__ o_in,
    const _Float16* __restrict__ w0f, const float* __restrict__ b0,
    const float* __restrict__ g0, const float* __restrict__ be0,
    const _Float16* __restrict__ w1f, const float* __restrict__ b1,
    const float* __restrict__ g1, const float* __restrict__ be1,
    const float* __restrict__ gout, const float* __restrict__ bout,
    float* __restrict__ dout) {
  __shared__ __align__(16) _Float16 ldsA[4][16 * 136];
  __shared__ __align__(16) _Float16 ldsB[4][16 * 264];
  const int wave = threadIdx.x >> 5;
  const int lane = threadIdx.x & 31;
  const int half = lane >> 4;
  const int col  = lane & 15;
  const int tile = blockIdx.x * 4 + wave;  // 576 tiles
  const int n0   = tile * 16;

  _Float16* A  = ldsA[wave];
  _Float16* Bh = ldsB[wave];
  float*    Sf = (float*)ldsB[wave];

  // stage 16x128 fp32 rows (contiguous) as f16
#pragma unroll 4
  for (int it = 0; it < 16; ++it) {
    float4 v = *(const float4*)(o_in + (size_t)(n0 + it) * 128 + lane * 4);
    union { _Float16 h[4]; uint2 u; } t4;
    t4.h[0] = (_Float16)v.x; t4.h[1] = (_Float16)v.y;
    t4.h[2] = (_Float16)v.z; t4.h[3] = (_Float16)v.w;
    *(uint2*)(A + it * 136 + lane * 4) = t4.u;
  }

  v16h afr[4];
#pragma unroll
  for (int kc = 0; kc < 4; ++kc) afr[kc] = lds_a_frag(A, 136, kc * 32, lane);

  // layer m0: 128 -> 256, GN4 (64-col groups), relu
#pragma unroll
  for (int g = 0; g < 4; ++g) {
    v8f acc[4];
#pragma unroll
    for (int jj = 0; jj < 4; ++jj) {
      int j = g * 4 + jj;
      v8f c8 = {0.f, 0.f, 0.f, 0.f, 0.f, 0.f, 0.f, 0.f};
#pragma unroll
      for (int kc = 0; kc < 4; ++kc)
        c8 = wmma16(afr[kc], glb_b_frag(w0f, 128, j * 16, kc * 32, lane), c8);
      float bias = b0[j * 16 + col];
#pragma unroll
      for (int e = 0; e < 8; ++e) c8[e] += bias;
      acc[jj] = c8;
    }
    float s[8], q[8];
#pragma unroll
    for (int e = 0; e < 8; ++e) {
      float sv = 0.f, qv = 0.f;
#pragma unroll
      for (int jj = 0; jj < 4; ++jj) { float v = acc[jj][e]; sv += v; qv += v * v; }
      s[e] = sv; q[e] = qv;
    }
    reduce_cols<8>(s, q);
#pragma unroll
    for (int jj = 0; jj < 4; ++jj) {
      int cch = (g * 4 + jj) * 16 + col;
      float gam = g0[cch], bet = be0[cch];
#pragma unroll
      for (int e = 0; e < 8; ++e) {
        float mean = s[e] * (1.f / 64.f);
        float var  = q[e] * (1.f / 64.f) - mean * mean;
        float y = (acc[jj][e] - mean) * rsqrtf(var + GN_EPS) * gam + bet;
        Bh[(e + 8 * half) * 264 + cch] = (_Float16)fmaxf(y, 0.f);
      }
    }
  }

  v16h af2[8];
#pragma unroll
  for (int kc = 0; kc < 8; ++kc) af2[kc] = lds_a_frag(Bh, 264, kc * 32, lane);

  // layer m1: 256 -> 128, GN4 (32-col groups), relu; keep in registers
  float fr[8][8];
#pragma unroll
  for (int g = 0; g < 4; ++g) {
    v8f acc[2];
#pragma unroll
    for (int jj = 0; jj < 2; ++jj) {
      int j = g * 2 + jj;
      v8f c8 = {0.f, 0.f, 0.f, 0.f, 0.f, 0.f, 0.f, 0.f};
#pragma unroll
      for (int kc = 0; kc < 8; ++kc)
        c8 = wmma16(af2[kc], glb_b_frag(w1f, 256, j * 16, kc * 32, lane), c8);
      float bias = b1[j * 16 + col];
#pragma unroll
      for (int e = 0; e < 8; ++e) c8[e] += bias;
      acc[jj] = c8;
    }
    float s[8], q[8];
#pragma unroll
    for (int e = 0; e < 8; ++e) {
      float v0 = acc[0][e], v1 = acc[1][e];
      s[e] = v0 + v1;
      q[e] = v0 * v0 + v1 * v1;
    }
    reduce_cols<8>(s, q);
#pragma unroll
    for (int jj = 0; jj < 2; ++jj) {
      int j = g * 2 + jj;
      float gam = g1[j * 16 + col], bet = be1[j * 16 + col];
#pragma unroll
      for (int e = 0; e < 8; ++e) {
        float mean = s[e] * (1.f / 32.f);
        float var  = q[e] * (1.f / 32.f) - mean * mean;
        float y = (acc[jj][e] - mean) * rsqrtf(var + GN_EPS) * gam + bet;
        fr[j][e] = fmaxf(y, 0.f);
      }
    }
  }

  // out_norm GN16: per-row stats over 8-column groups (aligned 8-lane groups)
#pragma unroll
  for (int j = 0; j < 8; ++j) {
    float s[8], q[8];
#pragma unroll
    for (int e = 0; e < 8; ++e) { s[e] = fr[j][e]; q[e] = fr[j][e] * fr[j][e]; }
    reduce_cols<4>(s, q);  // masks 1,2,4 stay inside the 8-col group
    int cch = j * 16 + col;
    float gam = gout[cch], bet = bout[cch];
#pragma unroll
    for (int e = 0; e < 8; ++e) {
      float mean = s[e] * (1.f / 8.f);
      float var  = q[e] * (1.f / 8.f) - mean * mean;
      Sf[(e + 8 * half) * 132 + cch] =
          (fr[j][e] - mean) * rsqrtf(var + GN_EPS) * gam + bet;
    }
  }

  // scatter to (B,128,48,48): 16 consecutive n share b and pixel base p0
  int bb = n0 / 2304, p0 = n0 % 2304;
#pragma unroll 4
  for (int it = 0; it < 64; ++it) {
    int idx = it * 32 + lane;
    int c = idx >> 4, r = idx & 15;
    dout[(size_t)bb * 294912 + (size_t)c * 2304 + p0 + r] = Sf[r * 132 + c];
  }
}

// ---------------------------------------------------------------------------
extern "C" void kernel_launch(void* const* d_in, const int* in_sizes, int n_in,
                              void* d_out, int out_size, void* d_ws,
                              size_t ws_size, hipStream_t stream) {
  (void)in_sizes; (void)n_in; (void)out_size; (void)ws_size;
  const float* x        = (const float*)d_in[0];
  const int*   bpos     = (const int*)d_in[1];
  const float* w_in0    = (const float*)d_in[2];
  const float* b_in0    = (const float*)d_in[3];
  const float* g_in0    = (const float*)d_in[4];
  const float* be_in0   = (const float*)d_in[5];
  const float* w_in1    = (const float*)d_in[6];
  const float* b_in1    = (const float*)d_in[7];
  const float* g_in1    = (const float*)d_in[8];
  const float* be_in1   = (const float*)d_in[9];
  const float* g_innorm = (const float*)d_in[10];
  const float* b_innorm = (const float*)d_in[11];
  const float* Q        = (const float*)d_in[12];
  const float* w_k      = (const float*)d_in[13];
  const float* b_k      = (const float*)d_in[14];
  const float* w_m0     = (const float*)d_in[15];
  const float* b_m0     = (const float*)d_in[16];
  const float* g_m0     = (const float*)d_in[17];
  const float* be_m0    = (const float*)d_in[18];
  const float* w_m1     = (const float*)d_in[19];
  const float* b_m1     = (const float*)d_in[20];
  const float* g_m1     = (const float*)d_in[21];
  const float* be_m1    = (const float*)d_in[22];
  const float* g_out    = (const float*)d_in[23];
  const float* b_out    = (const float*)d_in[24];

  // workspace layout
  char* ws = (char*)d_ws;
  float*    ws1 = (float*)ws;                              // 221184*128*4 B
  float*    ws2 = (float*)(ws + 113246208);                // 9216*128*4 B
  _Float16* wf  = (_Float16*)(ws + 113246208 + 4718592);   // 4*32768 f16

  cvt_w<<<512, 256, 0, stream>>>(w_in0, w_in1, w_m0, w_m1, wf);
  ltae_inconv<<<3456, 128, 0, stream>>>(x, wf, b_in0, g_in0, be_in0,
                                        wf + 32768, b_in1, g_in1, be_in1, ws1);
  ltae_attn<<<9216, 256, 0, stream>>>(ws1, bpos, g_innorm, b_innorm, Q, w_k,
                                      b_k, ws2);
  ltae_outmlp<<<144, 128, 0, stream>>>(ws2, wf + 65536, b_m0, g_m0, be_m0,
                                       wf + 98304, b_m1, g_m1, be_m1, g_out,
                                       b_out, (float*)d_out);
}
